// VM_candidate_model_43696997269786
// MI455X (gfx1250) — matrix operand order repaired and checked by reading, not compiled, not run on hardware
//
#include <hip/hip_runtime.h>
#include <hip/hip_bf16.h>

#define NUM_PM 254
#define NUM_VM 768
#define SEQQ   (NUM_PM + NUM_VM + 2)   /* 1024 */
#define HH     8
#define DD     256
#define DHH    32
#define FFN    1024
#define LLAY   3
#define BBB    8

typedef __bf16 bf16;
typedef __attribute__((ext_vector_type(16))) __bf16 v16bf;
typedef __attribute__((ext_vector_type(8)))  float  v8f;

#if defined(__has_builtin)
#if __has_builtin(__builtin_amdgcn_s_wait_asynccnt)
#define HAVE_ASYNC_LDS 1
#endif
#endif

// ---------------- wave32 reductions ----------------
static __device__ __forceinline__ float wmax16(float v) {
  v = fmaxf(v, __shfl_xor(v, 1)); v = fmaxf(v, __shfl_xor(v, 2));
  v = fmaxf(v, __shfl_xor(v, 4)); v = fmaxf(v, __shfl_xor(v, 8));
  return v;
}
static __device__ __forceinline__ float wsum16(float v) {
  v += __shfl_xor(v, 1); v += __shfl_xor(v, 2);
  v += __shfl_xor(v, 4); v += __shfl_xor(v, 8);
  return v;
}
static __device__ __forceinline__ float wsum32(float v) {
  v += __shfl_xor(v, 1); v += __shfl_xor(v, 2);
  v += __shfl_xor(v, 4); v += __shfl_xor(v, 8);
  v += __shfl_xor(v, 16);
  return v;
}

// ---------------- weight transpose + f32->bf16 (dst is N x K, contiguous in K) ----
__global__ void tconv_kernel(const float* __restrict__ src, bf16* __restrict__ dst,
                             int K, int N) {
  int idx = blockIdx.x * blockDim.x + threadIdx.x;
  if (idx >= K * N) return;
  int n = idx / K, k = idx % K;
  dst[idx] = (bf16)src[(size_t)k * N + n];
}

// ---------------- embedding: build x[b, s, :] -------------------------------
__global__ void embed_kernel(const float* __restrict__ vm_states,
                             const float* __restrict__ num_step,
                             const float* __restrict__ pm_states,
                             const float* __restrict__ pm_W, const float* __restrict__ pm_b,
                             const float* __restrict__ vm_W, const float* __restrict__ vm_b,
                             float* __restrict__ x) {
  int s = blockIdx.x % SEQQ;
  int b = blockIdx.x / SEQQ;
  int d = threadIdx.x;               // 256 threads
  float val;
  if (s == 0) {
    val = num_step[b];
  } else if (s <= NUM_PM) {
    const float* p = pm_states + ((size_t)b * NUM_PM + (s - 1)) * 16;
    float acc = pm_b[d];
#pragma unroll
    for (int k = 0; k < 16; ++k) acc += p[k] * pm_W[k * DD + d];
    val = acc;
  } else if (s < SEQQ - 1) {
    const float* p = vm_states + ((size_t)b * NUM_VM + (s - 1 - NUM_PM)) * 16;
    float acc = vm_b[d];
#pragma unroll
    for (int k = 0; k < 16; ++k) acc += p[k] * vm_W[k * DD + d];
    val = acc;
  } else {
    val = -1.0f;
  }
  x[((size_t)b * SEQQ + s) * DD + d] = val;
}

// ---------------- LayerNorm: wave per row, D=256 (8 elems/lane) -------------
__global__ void ln_kernel(const float* __restrict__ x,
                          const float* __restrict__ g, const float* __restrict__ bta,
                          bf16* __restrict__ h) {
  int wave = threadIdx.x >> 5;
  int lane = threadIdx.x & 31;
  int row  = blockIdx.x * (blockDim.x >> 5) + wave;
  const float* xr = x + (size_t)row * DD;
  float v[8];
  float s = 0.f;
#pragma unroll
  for (int i = 0; i < 8; ++i) { v[i] = xr[lane + 32 * i]; s += v[i]; }
  s = wsum32(s);
  float mean = s * (1.0f / DD);
  float var = 0.f;
#pragma unroll
  for (int i = 0; i < 8; ++i) { float d = v[i] - mean; var += d * d; }
  var = wsum32(var);
  float rstd = rsqrtf(var * (1.0f / DD) + 1e-5f);
  bf16* hr = h + (size_t)row * DD;
#pragma unroll
  for (int i = 0; i < 8; ++i) {
    int c = lane + 32 * i;
    hr[c] = (bf16)((v[i] - mean) * rstd * g[c] + bta[c]);
  }
}

// ---------------- blocked WMMA GEMM ------------------------------------------
// C[M,N] = A[M,K](bf16, row major) * Wt[N,K](bf16)^T + bias
// Workgroup: 8 waves. WG tile: 256(M) x 64(N). Wave tile: 32(M) x 64(N).
// The 64xGK weight slice is staged per-k-chunk in LDS (async-to-LDS via inline
// asm: GLOBAL_LOAD_ASYNC_TO_LDS_B128, ISA 15.18.3 op 98) and reused by 8 waves.
// flags: bit0 = gelu(tanh) epilogue; bit1 = residual add into/out of resid(f32,ld=N)
#define GK  64   /* k chunk */
#define GNB 64   /* N per workgroup */
__global__ __launch_bounds__(256) void gemm_wmma_kernel(
    const bf16* __restrict__ A, int lda,
    const bf16* __restrict__ Wt,          // N x K
    const float* __restrict__ bias,
    float* __restrict__ resid,            // nullable (ld = N)
    bf16* __restrict__ outb,              // nullable (ld = N)
    int N, int K, int flags) {
  // NOTE: Wtile is the only LDS object in this kernel -> its LDS base offset is 0,
  // so a lane's LDS byte address for the async copy is just its element offset.
  __shared__ alignas(64) bf16 Wtile[GNB * GK];  // 8 KB
  int tid  = threadIdx.x;
  int lane = tid & 31;
  int wave = tid >> 5;
  int lo = lane & 15, hi = lane >> 4;
  int tn  = blockIdx.x * GNB;
  int tm  = blockIdx.y * (8 * 32) + wave * 32;

  v8f acc[2][4];
#pragma unroll
  for (int a = 0; a < 2; ++a)
#pragma unroll
    for (int j = 0; j < 4; ++j) acc[a][j] = (v8f){};

  const bf16* a0p = A + (size_t)(tm + lo) * lda + hi * 16;
  const bf16* a1p = A + (size_t)(tm + 16 + lo) * lda + hi * 16;

  for (int k0 = 0; k0 < K; k0 += GK) {
    // ---- stage Wt[tn..tn+63][k0..k0+63] into LDS (4096 bf16, 16 per thread)
    {
      int e  = tid * 16;                      // element index in Wtile
      int n  = e >> 6;                        // / GK
      int kk = e & (GK - 1);
      const bf16* g = Wt + (size_t)(tn + n) * K + k0 + kk;
#if defined(HAVE_ASYNC_LDS)
      unsigned loff = (unsigned)(e * (int)sizeof(bf16));   // LDS byte address
      unsigned long long ga = (unsigned long long)g;
      // INST_OFFSET is added to BOTH the LDS and the global address (ISA 10.7),
      // so offset:16 moves the second 16-byte half of this lane's 32B chunk.
      // Wtile is passed as an (unreferenced) operand so its address escapes and
      // the "memory" clobber forces later LDS reads to see the async writes.
      asm volatile("global_load_async_to_lds_b128 %0, %1, off"
                   :: "v"(loff), "v"(ga), "v"((const void*)Wtile) : "memory");
      asm volatile("global_load_async_to_lds_b128 %0, %1, off offset:16"
                   :: "v"(loff), "v"(ga), "v"((const void*)Wtile) : "memory");
      __builtin_amdgcn_s_wait_asynccnt(0);
#else
      *(v16bf*)(Wtile + e) = *(const v16bf*)g;
#endif
    }
    __syncthreads();
    // ---- 32x64 wave tile: 8 WMMAs per 32-wide k step
#pragma unroll
    for (int kk = 0; kk < GK; kk += 32) {
      v16bf a0 = *(const v16bf*)(a0p + k0 + kk);
      v16bf a1 = *(const v16bf*)(a1p + k0 + kk);
#pragma unroll
      for (int j = 0; j < 4; ++j) {
        v16bf bfr = *(const v16bf*)(Wtile + (j * 16 + lo) * GK + kk + hi * 16);
        acc[0][j] = __builtin_amdgcn_wmma_f32_16x16x32_bf16(false, a0, false, bfr,
                                                            (short)0, acc[0][j], false, false);
        acc[1][j] = __builtin_amdgcn_wmma_f32_16x16x32_bf16(false, a1, false, bfr,
                                                            (short)0, acc[1][j], false, false);
      }
    }
    __syncthreads();
  }

  // ---- epilogue
#pragma unroll
  for (int a = 0; a < 2; ++a) {
#pragma unroll
    for (int j = 0; j < 4; ++j) {
      int n = tn + j * 16 + lo;
      float bv = bias ? bias[n] : 0.f;
#pragma unroll
      for (int i = 0; i < 8; ++i) {
        int m = tm + a * 16 + i + hi * 8;
        float v = acc[a][j][i] + bv;
        if (flags & 1) {               // tanh-approx gelu
          float u = v;
          float c = 0.7978845608028654f * (u + 0.044715f * u * u * u);
          v = 0.5f * u * (1.0f + tanhf(c));
        }
        size_t o = (size_t)m * N + n;
        if (flags & 2) { v += resid[o]; resid[o] = v; }
        if (outb) outb[o] = (bf16)v;
      }
    }
  }
}

// ---------------- masked flash attention: wave per (b, head, 16-row q tile) --
static __device__ __forceinline__ int key_id(const int* __restrict__ rel, int b, int col) {
  if (col == 0) return -2;
  if (col == SEQQ - 1) return -3;
  return rel[(size_t)b * (SEQQ - 2) + col - 1];
}
static __device__ __forceinline__ bool key_pad(const unsigned char* __restrict__ msk,
                                               int b, int col) {
  int v = col - (1 + NUM_PM);
  return (v >= 0 && v < NUM_VM) ? (msk[(size_t)b * NUM_VM + v] != 0) : false;
}

__global__ void attn_kernel(const bf16* __restrict__ qkv,   // [B*SEQ, 3*D] bf16
                            const int* __restrict__ rel,
                            const unsigned char* __restrict__ vmmask,
                            bf16* __restrict__ o) {          // [B*SEQ, D] bf16
  int qt = blockIdx.x, hd = blockIdx.y, b = blockIdx.z;
  int lane = threadIdx.x;
  int lo = lane & 15, hi = lane >> 4;
  int qbase = qt * 16;
  const float scale = 0.17677669529663687f;  // 1/sqrt(32)
  __shared__ alignas(64) bf16 Pst[16 * 32];

  // Q fragment: 16x32 bf16
  const bf16* qrow = qkv + ((size_t)(b * SEQQ + qbase + lo)) * (3 * DD) + hd * DHH + hi * 16;
  v16bf qa = *(const v16bf*)qrow;

  int rq[8];
#pragma unroll
  for (int i = 0; i < 8; ++i) rq[i] = key_id(rel, b, qbase + i + hi * 8);

  float mrow[8], lrow[8];
  v8f O0 = {}, O1 = {};
#pragma unroll
  for (int i = 0; i < 8; ++i) { mrow[i] = -1e30f; lrow[i] = 0.f; }

  for (int kb = 0; kb < SEQQ; kb += 32) {
    // K^T fragments for two 16-key halves
    const bf16* kp0 = qkv + ((size_t)(b * SEQQ + kb + lo)) * (3 * DD) + DD + hd * DHH + hi * 16;
    v16bf k0 = *(const v16bf*)kp0;
    v16bf k1 = *(const v16bf*)(kp0 + (size_t)16 * (3 * DD));
    v8f s0 = {}, s1 = {};
    s0 = __builtin_amdgcn_wmma_f32_16x16x32_bf16(false, qa, false, k0, (short)0, s0, false, false);
    s1 = __builtin_amdgcn_wmma_f32_16x16x32_bf16(false, qa, false, k1, (short)0, s1, false, false);

    int  rk0 = key_id(rel, b, kb + lo),        rk1 = key_id(rel, b, kb + 16 + lo);
    bool pd0 = key_pad(vmmask, b, kb + lo),    pd1 = key_pad(vmmask, b, kb + 16 + lo);

#pragma unroll
    for (int i = 0; i < 8; ++i) {
      float e0 = s0[i] * scale + (((rq[i] != rk0) || pd0) ? -1e9f : 0.f);
      float e1 = s1[i] * scale + (((rq[i] != rk1) || pd1) ? -1e9f : 0.f);
      float mx = wmax16(fmaxf(e0, e1));
      float mnew = fmaxf(mrow[i], mx);
      float corr = __expf(mrow[i] - mnew);
      float p0 = __expf(e0 - mnew);
      float p1 = __expf(e1 - mnew);
      lrow[i] = lrow[i] * corr + wsum16(p0 + p1);
      O0[i] *= corr; O1[i] *= corr;
      mrow[i] = mnew;
      int m = i + hi * 8;
      Pst[m * 32 + lo]      = (bf16)p0;
      Pst[m * 32 + lo + 16] = (bf16)p1;
    }
    __syncthreads();
    v16bf pa = *(const v16bf*)(Pst + lo * 32 + hi * 16);   // A-layout reload via LDS
    const bf16* vp = qkv + ((size_t)(b * SEQQ + kb + hi * 16)) * (3 * DD) + 2 * DD + hd * DHH + lo;
    v16bf v0, v1;
#pragma unroll
    for (int s = 0; s < 16; ++s) {
      v0[s] = vp[(size_t)s * (3 * DD)];
      v1[s] = vp[(size_t)s * (3 * DD) + 16];
    }
    O0 = __builtin_amdgcn_wmma_f32_16x16x32_bf16(false, pa, false, v0, (short)0, O0, false, false);
    O1 = __builtin_amdgcn_wmma_f32_16x16x32_bf16(false, pa, false, v1, (short)0, O1, false, false);
    __syncthreads();
  }

#pragma unroll
  for (int i = 0; i < 8; ++i) {
    float inv = 1.0f / lrow[i];
    int q = qbase + i + hi * 8;
    bf16* orow = o + ((size_t)(b * SEQQ + q)) * DD + hd * DHH;
    orow[lo]      = (bf16)(O0[i] * inv);
    orow[lo + 16] = (bf16)(O1[i] * inv);
  }
}

// ---------------- output head: wave per output scalar -----------------------
__global__ void head_kernel(const float* __restrict__ x,
                            const float* __restrict__ outW, const float* __restrict__ outB,
                            const float* __restrict__ critW, const float* __restrict__ critB,
                            float* __restrict__ out) {
  int idx = blockIdx.x;
  int b = idx / (NUM_VM + 1);
  int j = idx % (NUM_VM + 1);
  int lane = threadIdx.x;
  bool crit = (j == NUM_VM);
  int row = crit ? (SEQQ - 1) : (NUM_PM + 1 + j);
  const float* w  = crit ? critW : outW;
  const float* xr = x + ((size_t)b * SEQQ + row) * DD;
  float s = 0.f;
#pragma unroll
  for (int i = 0; i < 8; ++i) s += xr[lane + 32 * i] * w[lane + 32 * i];
  s = wsum32(s);
  if (lane == 0) out[(size_t)b * (NUM_VM + 1) + j] = s + (crit ? critB[0] : outB[0]);
}

// ---------------- host launcher ---------------------------------------------
extern "C" void kernel_launch(void* const* d_in, const int* in_sizes, int n_in,
                              void* d_out, int out_size, void* d_ws, size_t ws_size,
                              hipStream_t stream) {
  const float* vm_states = (const float*)d_in[0];
  const float* num_step  = (const float*)d_in[1];
  const float* pm_states = (const float*)d_in[2];
  const int*   rel       = (const int*)d_in[3];
  const unsigned char* vmmask = (const unsigned char*)d_in[4];
  const float* pm_W = (const float*)d_in[5];
  const float* pm_b = (const float*)d_in[6];
  const float* vm_W = (const float*)d_in[7];
  const float* vm_b = (const float*)d_in[8];
  const float* Wqkv = (const float*)d_in[9];
  const float* bqkv = (const float*)d_in[10];
  const float* Wo   = (const float*)d_in[11];
  const float* bo   = (const float*)d_in[12];
  const float* ln1g = (const float*)d_in[13];
  const float* ln1b = (const float*)d_in[14];
  const float* ln2g = (const float*)d_in[15];
  const float* ln2b = (const float*)d_in[16];
  const float* W1   = (const float*)d_in[17];
  const float* b1   = (const float*)d_in[18];
  const float* W2   = (const float*)d_in[19];
  const float* b2   = (const float*)d_in[20];
  const float* outW = (const float*)d_in[21];
  const float* outB = (const float*)d_in[22];
  const float* critW = (const float*)d_in[23];
  const float* critB = (const float*)d_in[24];
  float* out = (float*)d_out;

  char* wsb = (char*)d_ws;
  size_t off = 0;
  auto wsalloc = [&](size_t bytes) -> void* {
    void* p = wsb + off;
    off += (bytes + 255) & ~(size_t)255;
    return p;
  };
  const int M = BBB * SEQQ;                                  // 8192 rows
  float* x    = (float*)wsalloc((size_t)M * DD * 4);
  bf16* h     = (bf16*)wsalloc((size_t)M * DD * 2);
  bf16* qkvb  = (bf16*)wsalloc((size_t)M * 3 * DD * 2);
  bf16* ob    = (bf16*)wsalloc((size_t)M * DD * 2);
  bf16* ffn   = (bf16*)wsalloc((size_t)M * FFN * 2);
  bf16* wqkvT = (bf16*)wsalloc((size_t)LLAY * 3 * DD * DD * 2);
  bf16* woT   = (bf16*)wsalloc((size_t)LLAY * DD * DD * 2);
  bf16* w1T   = (bf16*)wsalloc((size_t)LLAY * FFN * DD * 2);
  bf16* w2T   = (bf16*)wsalloc((size_t)LLAY * DD * FFN * 2);

  // 1) weight transpose+convert (K x N f32 -> N x K bf16)
  for (int i = 0; i < LLAY; ++i) {
    int n;
    n = DD * 3 * DD;
    tconv_kernel<<<(n + 255) / 256, 256, 0, stream>>>(Wqkv + (size_t)i * n, wqkvT + (size_t)i * n, DD, 3 * DD);
    n = DD * DD;
    tconv_kernel<<<(n + 255) / 256, 256, 0, stream>>>(Wo + (size_t)i * n, woT + (size_t)i * n, DD, DD);
    n = DD * FFN;
    tconv_kernel<<<(n + 255) / 256, 256, 0, stream>>>(W1 + (size_t)i * n, w1T + (size_t)i * n, DD, FFN);
    tconv_kernel<<<(n + 255) / 256, 256, 0, stream>>>(W2 + (size_t)i * n, w2T + (size_t)i * n, FFN, DD);
  }

  // 2) embedding
  embed_kernel<<<BBB * SEQQ, 256, 0, stream>>>(vm_states, num_step, pm_states,
                                               pm_W, pm_b, vm_W, vm_b, x);

  // 3) transformer layers
  const int MB = M / 256;   // grid.y for gemm (256 rows per WG)
  for (int i = 0; i < LLAY; ++i) {
    ln_kernel<<<M / 8, 256, 0, stream>>>(x, ln1g + i * DD, ln1b + i * DD, h);
    gemm_wmma_kernel<<<dim3(3 * DD / GNB, MB), 256, 0, stream>>>(
        h, DD, wqkvT + (size_t)i * 3 * DD * DD, bqkv + (size_t)i * 3 * DD,
        nullptr, qkvb, 3 * DD, DD, 0);
    attn_kernel<<<dim3(SEQQ / 16, HH, BBB), 32, 0, stream>>>(qkvb, rel, vmmask, ob);
    gemm_wmma_kernel<<<dim3(DD / GNB, MB), 256, 0, stream>>>(
        ob, DD, woT + (size_t)i * DD * DD, bo + (size_t)i * DD,
        x, nullptr, DD, DD, 2);
    ln_kernel<<<M / 8, 256, 0, stream>>>(x, ln2g + i * DD, ln2b + i * DD, h);
    gemm_wmma_kernel<<<dim3(FFN / GNB, MB), 256, 0, stream>>>(
        h, DD, w1T + (size_t)i * FFN * DD, b1 + (size_t)i * FFN,
        nullptr, ffn, FFN, DD, 1);
    gemm_wmma_kernel<<<dim3(DD / GNB, MB), 256, 0, stream>>>(
        ffn, FFN, w2T + (size_t)i * DD * FFN, b2 + (size_t)i * DD,
        x, nullptr, DD, FFN, 2);
  }

  // 4) output head
  head_kernel<<<BBB * (NUM_VM + 1), 32, 0, stream>>>(x, outW, outB, critW, critB, out);
}
